// EquivariantNN_44856638440004
// MI455X (gfx1250) — compile-verified, matrix operand order checked by reading, not run on hardware
//
#include <hip/hip_runtime.h>

// CDNA5 (gfx1250) fused equivariant MLP: 9 -> 49 -> 49 -> 9, tied weights.
// All matmuls via v_wmma_f32_16x16x32_f16 (wave32). B matrices pre-packed in
// LDS in WMMA fragment order (2x ds_load_b128 per fragment); branchless
// global input load; relu done as packed f16 v_pk_max_num_f16 after cvt_pk.

typedef __attribute__((ext_vector_type(16))) _Float16 v16h;
typedef __attribute__((ext_vector_type(8)))  _Float16 v8h;
typedef __attribute__((ext_vector_type(2)))  __fp16   v2fp16;  // pkrtz result type
typedef __attribute__((ext_vector_type(8)))  float    v8f;

#define WAVES_PER_BLOCK 8
#define ROWS_PER_WAVE   16
#define ROWS_PER_BLOCK  (WAVES_PER_BLOCK * ROWS_PER_WAVE)
#define STG_STRIDE      72   // halves per staged row: 144 B (multiple of 16 B)

// Fragment indices in the packed-B LDS array:
//   0..3   : layer-1 B tiles, nt = f           (K-chunk 0 only, K<9)
//   4..11  : layer-2 B tiles, nt = (f-4)>>1, kc = (f-4)&1
//   12..13 : layer-3 B tiles, kc = f-12
#define NFRAGS 14

// K index within a 32-wide K-chunk for 16-bit A/B fragments (ISA 7.12.2):
// lanes 0-15 hold K = {0..7, 16..23}, lanes 16-31 hold K = {8..15, 24..31}.
__device__ __forceinline__ int kmap(int lane, int h) {
    int kb = (lane >> 4) << 3;                 // 0 or 8
    return (h < 8) ? (kb + h) : (16 + kb + (h - 8));
}

// Packed B-fragment load: 32 contiguous bytes per lane -> 2x ds_load_b128.
__device__ __forceinline__ v16h load_pb(const _Float16* base, int lane) {
    return *(const v16h*)(base + lane * 16);
}

// A fragment reload from the per-wave staging tile: two ds_load_b128 per lane.
__device__ __forceinline__ v16h load_afrag_stage(const _Float16* stg,
                                                 int lane, int kOff) {
    int row = lane & 15;
    int kb  = (lane >> 4) << 3;
    const v8h* lo = (const v8h*)(stg + row * STG_STRIDE + kOff + kb);
    const v8h* hi = (const v8h*)(stg + row * STG_STRIDE + kOff + 16 + kb);
    v8h a0 = *lo;
    v8h a1 = *hi;
    return __builtin_shufflevector(a0, a1, 0, 1, 2, 3, 4, 5, 6, 7,
                                   8, 9, 10, 11, 12, 13, 14, 15);
}

__device__ __forceinline__ v8f splat8(float v) {
    v8f c;
#pragma unroll
    for (int r = 0; r < 8; ++r) c[r] = v;
    return c;
}

// Pack 2x f32 -> packed f16 (v_cvt_pk_rtz_f16_f32), then relu both halves in
// one packed op (v_pk_max_num_f16 with inline-0 replicated to both halves).
__device__ __forceinline__ void store_tile_relu(_Float16* stg, int mBase, int n,
                                                v8f acc) {
#pragma unroll
    for (int j = 0; j < 4; ++j) {
        v2fp16 pr = __builtin_amdgcn_cvt_pkrtz(acc[2 * j], acc[2 * j + 1]);
        asm("v_pk_max_num_f16 %0, %1, 0" : "=v"(pr) : "v"(pr));
        stg[(mBase + 2 * j)     * STG_STRIDE + n] = (_Float16)pr[0];
        stg[(mBase + 2 * j + 1) * STG_STRIDE + n] = (_Float16)pr[1];
    }
}

__global__ __launch_bounds__(256)
void equivariant_mlp_kernel(const float* __restrict__ x,
                            const int* __restrict__ wp1, const int* __restrict__ bp1,
                            const int* __restrict__ wp2, const int* __restrict__ bp2,
                            const int* __restrict__ wp3, const int* __restrict__ bp3,
                            const float* __restrict__ w1, const float* __restrict__ b1,
                            const float* __restrict__ w2, const float* __restrict__ b2,
                            const float* __restrict__ w3, const float* __restrict__ b3,
                            float* __restrict__ out, int batch) {
    __shared__ __attribute__((aligned(32))) _Float16 pB[NFRAGS][32 * 16];
    __shared__ __attribute__((aligned(16)))
        _Float16 stageAll[WAVES_PER_BLOCK][ROWS_PER_WAVE * STG_STRIDE];

    const int tid = threadIdx.x;

    // ---- build the 14 B fragments directly in WMMA fragment order ----
    for (int i = tid; i < NFRAGS * 512; i += 256) {
        int f  = i >> 9;          // fragment
        int e  = i & 511;         // entry within fragment
        int lf = e >> 4;          // destination lane
        int h  = e & 15;          // half index within lane's v16h
        int col = lf & 15;
        float v = 0.f;
        if (f < 4) {                               // layer 1, nt = f
            int n = f * 16 + col;
            int K = kmap(lf, h);
            if (K < 9 && n < 49) { int p = wp1[K * 49 + n]; if (p > 0) v = w1[p - 1]; }
        } else if (f < 12) {                       // layer 2
            int g = f - 4;
            int nt = g >> 1, kc = g & 1;
            int n = nt * 16 + col;
            int K = kc * 32 + kmap(lf, h);
            if (K < 49 && n < 49) { int p = wp2[K * 49 + n]; if (p > 0) v = w2[p - 1]; }
        } else {                                   // layer 3, kc = f-12
            int kc = f - 12;
            int n = col;
            int K = kc * 32 + kmap(lf, h);
            if (K < 49 && n < 9) { int p = wp3[K * 9 + n]; if (p > 0) v = w3[p - 1]; }
        }
        pB[f][e] = (_Float16)v;
    }
    __syncthreads();

    const int lane  = tid & 31;
    const int wave  = tid >> 5;
    const int nlane = lane & 15;                 // N (or row for A frags)
    const int mBase = (lane >> 4) << 3;          // D rows: mBase + r
    const bool hiHalf = lane >= 16;
    _Float16* stg = stageAll[wave];

    // ---- per-thread bias scalars (tiny global gathers, L2 resident) ----
    float bv1[4], bv2[4];
#pragma unroll
    for (int nt = 0; nt < 4; ++nt) {
        int n = nt * 16 + nlane;
        float v = 0.f;
        if (n < 49) { int p = bp1[n]; if (p > 0) v = b1[p - 1]; }
        bv1[nt] = v;
        v = 0.f;
        if (n < 49) { int p = bp2[n]; if (p > 0) v = b2[p - 1]; }
        bv2[nt] = v;
    }
    float bv3 = 0.f;
    if (nlane < 9) { int p = bp3[nlane]; if (p > 0) bv3 = b3[p - 1]; }

    const int totalChunks = (batch + ROWS_PER_BLOCK - 1) / ROWS_PER_BLOCK;
    for (int chunk = blockIdx.x; chunk < totalChunks; chunk += gridDim.x) {
        const long rowBase = (long)chunk * ROWS_PER_BLOCK + (long)wave * ROWS_PER_WAVE;

        // ---------- layer 1: [16 x 9] x [9 x 49] (padded 16x32 x 32x64)
        // Branch-free input: every lane loads the same 9 contiguous (always
        // in-bounds) floats of its row, then builds the fragment with selects.
        v16h a1;
        {
            long rowg = rowBase + nlane;
            if (rowg >= batch) rowg = batch - 1;   // clamp (stores are guarded)
            const float* xr = x + rowg * 9;
            float vv[9];
#pragma unroll
            for (int c = 0; c < 9; ++c) vv[c] = xr[c];
            // lanes 0-15: K=h for h<8; lanes 16-31: only K=8 (at h=0) is valid.
            a1[0] = (_Float16)(hiHalf ? vv[8] : vv[0]);
#pragma unroll
            for (int h = 1; h < 8; ++h)
                a1[h] = hiHalf ? (_Float16)0.f : (_Float16)vv[h];
#pragma unroll
            for (int h = 8; h < 16; ++h) a1[h] = (_Float16)0.f;
        }
        {
            v8f acc[4];
#pragma unroll
            for (int nt = 0; nt < 4; ++nt) {
                v16h bf = load_pb(pB[nt], lane);
                acc[nt] = __builtin_amdgcn_wmma_f32_16x16x32_f16(
                    false, a1, false, bf, (short)0, splat8(bv1[nt]), false, false);
            }
#pragma unroll
            for (int nt = 0; nt < 4; ++nt)
                store_tile_relu(stg, mBase, nt * 16 + nlane, acc[nt]);
        }

        // ---------- layer 2: [16 x 49] x [49 x 49] (padded 16x64 x 64x64)
        v16h a20 = load_afrag_stage(stg, lane, 0);
        v16h a21 = load_afrag_stage(stg, lane, 32);
        {
            v8f acc[4];
#pragma unroll
            for (int nt = 0; nt < 4; ++nt) {
                v16h bf0 = load_pb(pB[4 + nt * 2 + 0], lane);
                v16h bf1 = load_pb(pB[4 + nt * 2 + 1], lane);
                v8f a = __builtin_amdgcn_wmma_f32_16x16x32_f16(
                    false, a20, false, bf0, (short)0, splat8(bv2[nt]), false, false);
                acc[nt] = __builtin_amdgcn_wmma_f32_16x16x32_f16(
                    false, a21, false, bf1, (short)0, a, false, false);
            }
#pragma unroll
            for (int nt = 0; nt < 4; ++nt)
                store_tile_relu(stg, mBase, nt * 16 + nlane, acc[nt]);
        }

        // ---------- layer 3: [16 x 49] x [49 x 9] (padded 16x64 x 64x16)
        v16h a30 = load_afrag_stage(stg, lane, 0);
        v16h a31 = load_afrag_stage(stg, lane, 32);
        {
            v16h bf0 = load_pb(pB[12], lane);
            v16h bf1 = load_pb(pB[13], lane);
            v8f acc = __builtin_amdgcn_wmma_f32_16x16x32_f16(
                false, a30, false, bf0, (short)0, splat8(bv3), false, false);
            acc = __builtin_amdgcn_wmma_f32_16x16x32_f16(
                false, a31, false, bf1, (short)0, acc, false, false);
            if (nlane < 9) {
                long m0 = rowBase + mBase;
                float* op = out + m0 * 9 + nlane;
                if (m0 + 8 <= batch) {               // full-tile fast path
#pragma unroll
                    for (int r = 0; r < 8; ++r) op[r * 9] = acc[r];
                } else {
#pragma unroll
                    for (int r = 0; r < 8; ++r)
                        if (m0 + r < batch) op[r * 9] = acc[r];
                }
            }
        }
    }
}

extern "C" void kernel_launch(void* const* d_in, const int* in_sizes, int n_in,
                              void* d_out, int out_size, void* d_ws, size_t ws_size,
                              hipStream_t stream) {
    (void)n_in; (void)out_size; (void)d_ws; (void)ws_size;
    const float* x   = (const float*)d_in[0];
    const int*   wp1 = (const int*)d_in[1];
    const int*   bp1 = (const int*)d_in[2];
    const int*   wp2 = (const int*)d_in[3];
    const int*   bp2 = (const int*)d_in[4];
    const int*   wp3 = (const int*)d_in[5];
    const int*   bp3 = (const int*)d_in[6];
    const float* w1  = (const float*)d_in[7];
    const float* b1  = (const float*)d_in[8];
    const float* w2  = (const float*)d_in[9];
    const float* b2  = (const float*)d_in[10];
    const float* w3  = (const float*)d_in[11];
    const float* b3  = (const float*)d_in[12];
    float* out = (float*)d_out;

    const int batch = in_sizes[0] / 9;
    const int totalChunks = (batch + ROWS_PER_BLOCK - 1) / ROWS_PER_BLOCK;
    int blocks = totalChunks < 2048 ? totalChunks : 2048;
    if (blocks < 1) blocks = 1;

    equivariant_mlp_kernel<<<dim3(blocks), dim3(256), 0, stream>>>(
        x, wp1, bp1, wp2, bp2, wp3, bp3, w1, b1, w2, b2, w3, b3, out, batch);
}
